// GraphSAGE_22771916604219
// MI455X (gfx1250) — matrix operand order repaired and checked by reading, not compiled, not run on hardware
//
#include <hip/hip_runtime.h>
#include <hip/hip_bf16.h>
#include <math.h>

// ---------------- problem constants (match reference) ----------------
#define NN   65536
#define BB   128
#define NP   512
#define EE   524288
#define DIN  128
#define CC   256
#define HH   4
#define DHH  64

typedef __attribute__((ext_vector_type(16))) __bf16 v16bf;
typedef __attribute__((ext_vector_type(8)))  float  v8f;
typedef __attribute__((ext_vector_type(4)))  unsigned int u32x4;
typedef __attribute__((ext_vector_type(8)))  int i32x8;
typedef __attribute__((ext_vector_type(4)))  int i32x4;

__device__ __forceinline__ v8f wmma_bf16(v16bf a, v16bf b, v8f c) {
    // D = A(16x32 bf16) x B(32x16 bf16) + C(16x16 f32)   -> v_wmma_f32_16x16x32_bf16
    return __builtin_amdgcn_wmma_f32_16x16x32_bf16(false, a, false, b, (short)0, c, false, false);
}

// Tensor Data Mover: 2D fp32 tile (tileRows x tileCols), row stride strideElems,
// global -> LDS at byte offset ldsOff. D# packed per CDNA5 ISA 8.3/8.4.
__device__ __forceinline__ void tdm_load_tile_f32(const float* gsrc, unsigned ldsOff,
                                                  int tileRows, int tileCols,
                                                  long strideElems) {
    unsigned long long ga = (unsigned long long)(const void*)gsrc;
    u32x4 g0;
    g0[0] = 1u;                                               // count=1, user descriptor
    g0[1] = ldsOff;                                           // lds_addr (bytes)
    g0[2] = (unsigned)(ga & 0xFFFFFFFFull);                   // global_addr[31:0]
    g0[3] = (unsigned)((ga >> 32) & 0x01FFFFFFull) | (2u << 30); // addr[56:32] | type=2
    unsigned d0 = (unsigned)tileCols;                         // tensor_dim0 == tile width
    unsigned d1 = (unsigned)tileRows;                         // tensor_dim1 == tile rows
    unsigned long long s0 = (unsigned long long)strideElems;  // tensor_dim0_stride
    i32x8 g1;
    g1[0] = (int)(2u << 16);                                  // wg_mask=0, data_size=4B
    g1[1] = (int)((d0 & 0xFFFFu) << 16);                      // tensor_dim0[15:0] @ 63:48
    g1[2] = (int)(((d0 >> 16) & 0xFFFFu) | ((d1 & 0xFFFFu) << 16));
    g1[3] = (int)(((d1 >> 16) & 0xFFFFu) | (((unsigned)tileCols & 0xFFFFu) << 16)); // tile_dim0
    g1[4] = (int)((unsigned)tileRows & 0xFFFFu);              // tile_dim1 (tile_dim2=0)
    g1[5] = (int)(s0 & 0xFFFFFFFFull);                        // dim0_stride[31:0]
    g1[6] = (int)((s0 >> 32) & 0xFFFFull);                    // dim0_stride[47:32]
    g1[7] = 0;
    i32x4 z4 = {0, 0, 0, 0};
#if defined(__clang_major__) && (__clang_major__ >= 23)
    i32x8 z8 = {0, 0, 0, 0, 0, 0, 0, 0};
    __builtin_amdgcn_tensor_load_to_lds(g0, g1, z4, z4, z8, 0);
#else
    __builtin_amdgcn_tensor_load_to_lds(g0, g1, z4, z4, 0);
#endif
}

__device__ __forceinline__ float gelu_f(float x) {
    return 0.5f * x * (1.0f + erff(x * 0.70710678118654752440f));
}

__device__ __forceinline__ float hredmax16(float x) {
    #pragma unroll
    for (int m = 1; m < 16; m <<= 1) x = fmaxf(x, __shfl_xor(x, m, 32));
    return x;
}
__device__ __forceinline__ float hredsum16(float x) {
    #pragma unroll
    for (int m = 1; m < 16; m <<= 1) x += __shfl_xor(x, m, 32);
    return x;
}

// ---------------- utility kernels ----------------
__global__ void k_zero(float* __restrict__ p, long n) {
    long i = (long)blockIdx.x * 256 + threadIdx.x;
    if (i < n) p[i] = 0.0f;
}

__global__ void k_degree(const int* __restrict__ dst, float* __restrict__ deg) {
    int e = blockIdx.x * 256 + threadIdx.x;
    if (e < EE) atomicAdd(&deg[dst[e]], 1.0f);
}

__global__ void k_rdeg(const float* __restrict__ deg, float* __restrict__ rdeg) {
    int i = blockIdx.x * 256 + threadIdx.x;
    if (i < NN) rdeg[i] = 1.0f / fmaxf(deg[i], 1.0f);
}

// scatter-add x[src] into agg[dst]; 4 channels per thread
__global__ void k_scatter(const float* __restrict__ x, const int* __restrict__ src,
                          const int* __restrict__ dst, float* __restrict__ agg, int K) {
    long gid = (long)blockIdx.x * 256 + threadIdx.x;
    int per = K >> 2;
    long e = gid / per;
    if (e >= EE) return;
    int c = (int)(gid % per) << 2;
    const float4 v = *(const float4*)(x + (long)src[e] * K + c);
    float* a = agg + (long)dst[e] * K + c;
    atomicAdd(a + 0, v.x);
    atomicAdd(a + 1, v.y);
    atomicAdd(a + 2, v.z);
    atomicAdd(a + 3, v.w);
}

// ---------------- generic dual GEMM with WMMA + TDM staging ----------------
// out[M][N] = (A1 .* rowscale1) @ W1^T  (+ A2 @ W2^T) (+ bias[N]) (+ addres[M][N])
// block: 256 thr = 8 waves; tile 32(M) x 128(N); grid (M/32, N/128).
// A staged 32x32 fp32 per k-step via tensor_load_to_lds (TENSORcnt).
__global__ __launch_bounds__(256) void k_gemm(
    const float* __restrict__ A1, const float* __restrict__ W1,
    const float* __restrict__ rowscale1,
    const float* __restrict__ A2, const float* __restrict__ W2,
    const float* __restrict__ bias, const float* __restrict__ addres,
    float* __restrict__ out, int M, int N, int K) {
    __shared__ float As[32][32];

    const int tid  = threadIdx.x;
    const int lane = tid & 31, wave = tid >> 5;
    const int half = lane >> 4, ml = lane & 15;
    const long m0 = (long)blockIdx.x * 32;
    const long n0 = (long)blockIdx.y * 128 + wave * 16;
    const unsigned ldsA = (unsigned)(size_t)(void*)&As[0][0];

    v8f acc[2];
    #pragma unroll
    for (int mi = 0; mi < 2; ++mi)
        #pragma unroll
        for (int v = 0; v < 8; ++v) acc[mi][v] = 0.f;

    #pragma unroll 1
    for (int pass = 0; pass < 2; ++pass) {
        const float* A  = pass ? A2 : A1;
        const float* W  = pass ? W2 : W1;
        const float* rs = pass ? nullptr : rowscale1;
        if (A == nullptr) continue;

        float scale[2];
        #pragma unroll
        for (int mi = 0; mi < 2; ++mi)
            scale[mi] = rs ? rs[m0 + mi * 16 + ml] : 1.0f;

        for (int k0 = 0; k0 < K; k0 += 32) {
            __syncthreads();
            if (wave == 0) {   // one TDM op per block per k-step
                tdm_load_tile_f32(A + m0 * K + k0, ldsA, 32, 32, K);
                __builtin_amdgcn_s_wait_tensorcnt(0);
            }
            __syncthreads();

            // A fragments: contiguous k = 16*half + s (same as B); fold rowscale + bf16 cvt here
            v16bf af[2];
            #pragma unroll
            for (int mi = 0; mi < 2; ++mi) {
                #pragma unroll
                for (int v = 0; v < 8; ++v) {
                    int kb = half * 16 + 2 * v;
                    const float2 p = *(const float2*)&As[mi * 16 + ml][kb];
                    af[mi][2 * v]     = (__bf16)(p.x * scale[mi]);
                    af[mi][2 * v + 1] = (__bf16)(p.y * scale[mi]);
                }
            }
            // B fragment straight from weights: B[k][n] = W[n][k]; lane=N, 16 contiguous K
            const float* wr = W + (n0 + ml) * K + k0 + half * 16;
            if (k0 + 32 < K) __builtin_prefetch(wr + 32, 0, 1);  // global_prefetch_b8
            const float4* w4 = (const float4*)wr;
            float4 wa = w4[0], wb = w4[1], wc = w4[2], wd = w4[3];
            v16bf bfv;
            bfv[0]  = (__bf16)wa.x; bfv[1]  = (__bf16)wa.y; bfv[2]  = (__bf16)wa.z; bfv[3]  = (__bf16)wa.w;
            bfv[4]  = (__bf16)wb.x; bfv[5]  = (__bf16)wb.y; bfv[6]  = (__bf16)wb.z; bfv[7]  = (__bf16)wb.w;
            bfv[8]  = (__bf16)wc.x; bfv[9]  = (__bf16)wc.y; bfv[10] = (__bf16)wc.z; bfv[11] = (__bf16)wc.w;
            bfv[12] = (__bf16)wd.x; bfv[13] = (__bf16)wd.y; bfv[14] = (__bf16)wd.z; bfv[15] = (__bf16)wd.w;

            acc[0] = wmma_bf16(af[0], bfv, acc[0]);
            acc[1] = wmma_bf16(af[1], bfv, acc[1]);
        }
    }

    // epilogue: D element v -> row mi*16 + v + 8*half, col = ml
    #pragma unroll
    for (int mi = 0; mi < 2; ++mi) {
        #pragma unroll
        for (int v = 0; v < 8; ++v) {
            long row = m0 + mi * 16 + v + 8 * half;
            long col = n0 + ml;
            float val = acc[mi][v];
            if (bias)   val += bias[col];
            if (addres) val += addres[row * N + col];
            out[row * N + col] = val;
        }
    }
}

// ---------------- SAGE post-ops ----------------
// L2-normalize each row of h (length C), one wave per row
__global__ void k_rownorm(float* __restrict__ h) {
    int row = blockIdx.x * 8 + (threadIdx.x >> 5);
    int lane = threadIdx.x & 31;
    float* r = h + (long)row * CC;
    float v[8]; float s = 0.f;
    #pragma unroll
    for (int j = 0; j < 8; ++j) { v[j] = r[lane + j * 32]; s += v[j] * v[j]; }
    #pragma unroll
    for (int m = 1; m < 32; m <<= 1) s += __shfl_xor(s, m, 32);
    float sc = 1.0f / fmaxf(sqrtf(s), 1e-12f);
    #pragma unroll
    for (int j = 0; j < 8; ++j) r[lane + j * 32] = v[j] * sc;
}

// per-graph per-channel mean (grid=B, block=C threads)
__global__ void k_gmean(const float* __restrict__ h, float* __restrict__ mean) {
    int g = blockIdx.x, c = threadIdx.x;
    const float* p = h + (long)g * NP * CC + c;
    float s = 0.f;
    for (int i = 0; i < NP; ++i) s += p[(long)i * CC];
    mean[g * CC + c] = s * (1.0f / NP);
}

// per-graph per-channel var of (h - alpha*mean)
__global__ void k_gvar(const float* __restrict__ h, const float* __restrict__ mean,
                       const float* __restrict__ alpha, float* __restrict__ var) {
    int g = blockIdx.x, c = threadIdx.x;
    float mu = alpha[c] * mean[g * CC + c];
    const float* p = h + (long)g * NP * CC + c;
    float s = 0.f;
    for (int i = 0; i < NP; ++i) { float d = p[(long)i * CC] - mu; s += d * d; }
    var[g * CC + c] = s * (1.0f / NP);
}

// out = gelu(w * (h - alpha*mean) * rsqrt(var+eps) + b) + addres
__global__ void k_gnapply(const float* __restrict__ h, const float* __restrict__ mean,
                          const float* __restrict__ var, const float* __restrict__ w,
                          const float* __restrict__ b, const float* __restrict__ alpha,
                          const float* __restrict__ addres, float* __restrict__ out) {
    long i = (long)blockIdx.x * 256 + threadIdx.x;
    int c = (int)(i & (CC - 1));
    int g = (int)((i >> 8) >> 9);
    float mu = alpha[c] * mean[g * CC + c];
    float cen = h[i] - mu;
    float val = gelu_f(w[c] * cen * rsqrtf(var[g * CC + c] + 1e-5f) + b[c]);
    if (addres) val += addres[i];
    out[i] = val;
}

// ---------------- flash attention (per graph, per head) ----------------
// q,k,v,o: [N][C] fp32, head h uses channels [h*64, h*64+64)
// grid = B*H*(NP/128); block = 256 thr (8 waves, 16 q-rows each)
__global__ __launch_bounds__(256) void k_attn(const float* __restrict__ q,
                                              const float* __restrict__ k,
                                              const float* __restrict__ v,
                                              float* __restrict__ o) {
    __shared__ __bf16 Kt[32][DHH];
    __shared__ __bf16 Vt[32][DHH];
    __shared__ __bf16 Ps[8][16][32];

    const int bid = blockIdx.x;
    const int qt = bid & 3, hh = (bid >> 2) & 3, g = bid >> 4;
    const int tid = threadIdx.x, lane = tid & 31, wave = tid >> 5;
    const int half = lane >> 4, ml = lane & 15;
    const long qrow0 = (long)g * NP + qt * 128 + wave * 16;
    const int hof = hh * DHH;
    const float sc = 0.125f;  // 1/sqrt(64)

    // Q fragments for dh chunks [0,32) and [32,64)
    v16bf qf[2];
    #pragma unroll
    for (int t = 0; t < 2; ++t) {
        #pragma unroll
        for (int vv = 0; vv < 8; ++vv) {
            int kb = t * 32 + half * 16 + 2 * vv;
            const float2 qq = *(const float2*)(q + (qrow0 + ml) * CC + hof + kb);
            qf[t][2 * vv]     = (__bf16)qq.x;
            qf[t][2 * vv + 1] = (__bf16)qq.y;
        }
    }

    v8f oacc[4];
    #pragma unroll
    for (int t = 0; t < 4; ++t)
        #pragma unroll
        for (int vv = 0; vv < 8; ++vv) oacc[t][vv] = 0.f;
    float mrow[8], lrow[8];
    #pragma unroll
    for (int vv = 0; vv < 8; ++vv) { mrow[vv] = -1e30f; lrow[vv] = 0.f; }

    for (int kt = 0; kt < NP; kt += 32) {
        __syncthreads();
        // stage 32x64 K and V tiles as bf16
        #pragma unroll
        for (int idx = tid; idx < 2048; idx += 256) {
            int r = idx >> 6, c = idx & 63;
            long node = (long)g * NP + kt + r;
            Kt[r][c] = (__bf16)k[node * CC + hof + c];
            Vt[r][c] = (__bf16)v[node * CC + hof + c];
        }
        __syncthreads();

        // S = Q @ K^T  (two 16-col tiles, K-dim = 64 -> 2 WMMA steps each)
        v8f sArr[2];
        #pragma unroll
        for (int nt = 0; nt < 2; ++nt) {
            v8f s = {0.f, 0.f, 0.f, 0.f, 0.f, 0.f, 0.f, 0.f};
            #pragma unroll
            for (int t = 0; t < 2; ++t) {
                v16bf bfv;
                #pragma unroll
                for (int e = 0; e < 16; ++e) bfv[e] = Kt[nt * 16 + ml][t * 32 + half * 16 + e];
                s = wmma_bf16(qf[t], bfv, s);
            }
            sArr[nt] = s;
        }

        // online softmax update + write P (bf16) to per-wave LDS
        #pragma unroll
        for (int vv = 0; vv < 8; ++vv) {
            float s0 = sArr[0][vv] * sc;
            float s1 = sArr[1][vv] * sc;
            float tmax = hredmax16(fmaxf(s0, s1));
            float mnew = fmaxf(mrow[vv], tmax);
            float corr = __expf(mrow[vv] - mnew);
            mrow[vv] = mnew;
            float p0 = __expf(s0 - mnew), p1 = __expf(s1 - mnew);
            lrow[vv] = lrow[vv] * corr + hredsum16(p0 + p1);
            #pragma unroll
            for (int t = 0; t < 4; ++t) oacc[t][vv] *= corr;
            Ps[wave][vv + 8 * half][ml]      = (__bf16)p0;
            Ps[wave][vv + 8 * half][16 + ml] = (__bf16)p1;
        }

        // P A-fragment (16x32) and O += P @ V
        v16bf pf;
        #pragma unroll
        for (int vv = 0; vv < 8; ++vv) {
            int kb = half * 16 + 2 * vv;
            pf[2 * vv]     = Ps[wave][ml][kb];
            pf[2 * vv + 1] = Ps[wave][ml][kb + 1];
        }
        #pragma unroll
        for (int t = 0; t < 4; ++t) {
            v16bf vf;
            #pragma unroll
            for (int e = 0; e < 16; ++e) vf[e] = Vt[half * 16 + e][t * 16 + ml];
            oacc[t] = wmma_bf16(pf, vf, oacc[t]);
        }
    }

    // normalize by row sums and write out
    #pragma unroll
    for (int t = 0; t < 4; ++t)
        #pragma unroll
        for (int vv = 0; vv < 8; ++vv) {
            long row = qrow0 + vv + 8 * half;
            o[row * CC + hof + t * 16 + ml] = oacc[t][vv] / lrow[vv];
        }
}

// ---------------- graph-mode LayerNorm ----------------
__global__ void k_lnstats(const float* __restrict__ x2, float* __restrict__ mu,
                          float* __restrict__ var) {
    __shared__ float rs[256], rss[256];
    int g = blockIdx.x, tid = threadIdx.x;
    const float* p = x2 + (long)g * NP * CC;
    float s = 0.f, ss = 0.f;
    for (int i = tid; i < NP * CC; i += 256) { float v = p[i]; s += v; ss += v * v; }
    rs[tid] = s; rss[tid] = ss;
    __syncthreads();
    for (int off = 128; off; off >>= 1) {
        if (tid < off) { rs[tid] += rs[tid + off]; rss[tid] += rss[tid + off]; }
        __syncthreads();
    }
    if (tid == 0) {
        float m = rs[0] * (1.0f / (NP * CC));
        mu[g] = m;
        var[g] = rss[0] * (1.0f / (NP * CC)) - m * m;
    }
}

__global__ void k_lnapply(const float* __restrict__ x2, const float* __restrict__ mu,
                          const float* __restrict__ var, const float* __restrict__ w,
                          const float* __restrict__ b, float* __restrict__ out) {
    long i = (long)blockIdx.x * 256 + threadIdx.x;
    int c = (int)(i & (CC - 1));
    int g = (int)(i >> 17);  // NP*CC = 2^17
    float val = (x2[i] - mu[g]) * rsqrtf(var[g] + 1e-5f) * w[c] + b[c];
    out[i] = gelu_f(val);
}

// ---------------- pooling (mean | max | sum) ----------------
__global__ void k_pool(const float* __restrict__ x3, float* __restrict__ pooled) {
    int g = blockIdx.x, c = threadIdx.x;
    const float* p = x3 + (long)g * NP * CC + c;
    float s = 0.f, mx = -1e30f;
    for (int i = 0; i < NP; ++i) { float v = p[(long)i * CC]; s += v; mx = fmaxf(mx, v); }
    pooled[g * 768 + c]       = s * (1.0f / NP);
    pooled[g * 768 + 256 + c] = mx;
    pooled[g * 768 + 512 + c] = s;
}

// ---------------- host orchestration ----------------
extern "C" void kernel_launch(void* const* d_in, const int* in_sizes, int n_in,
                              void* d_out, int out_size, void* d_ws, size_t ws_size,
                              hipStream_t stream) {
    const float* x        = (const float*)d_in[0];
    const int*   ei       = (const int*)d_in[1];
    const int*   srcI     = ei;
    const int*   dstI     = ei + EE;
    const float* l0_ll_w  = (const float*)d_in[3];
    const float* l0_ll_b  = (const float*)d_in[4];
    const float* l0_lr_w  = (const float*)d_in[5];
    const float* l0_res_w = (const float*)d_in[6];
    const float* lin_l_w  = (const float*)d_in[7];
    const float* lin_l_b  = (const float*)d_in[8];
    const float* lin_r_w  = (const float*)d_in[9];
    const float* gn_w     = (const float*)d_in[10];
    const float* gn_b     = (const float*)d_in[11];
    const float* gn_a     = (const float*)d_in[12];
    const float* ain_w    = (const float*)d_in[13];
    const float* ain_b    = (const float*)d_in[14];
    const float* aout_w   = (const float*)d_in[15];
    const float* aout_b   = (const float*)d_in[16];
    const float* ln_w     = (const float*)d_in[17];
    const float* ln_b     = (const float*)d_in[18];
    const float* pool_w   = (const float*)d_in[19];
    const float* pool_b   = (const float*)d_in[20];
    float* out = (float*)d_out;

    // workspace carve-up (~321 MB fp32)
    float* ws     = (float*)d_ws;
    float* deg    = ws;                       // N
    float* rdeg   = deg + NN;                 // N
    float* gmean  = rdeg + NN;                // B*C
    float* gvar   = gmean + BB * CC;          // B*C
    float* lnmu   = gvar + BB * CC;           // B
    float* lnvar  = lnmu + BB;                // B
    float* pooled = lnvar + BB;               // B*768
    float* big0   = pooled + (size_t)BB * 768;
    const size_t BIG = (size_t)NN * CC;
    float* big1 = big0 + BIG;
    float* big2 = big1 + BIG;
    float* big3 = big2 + BIG;
    float* big4 = big3 + BIG;
    (void)in_sizes; (void)n_in; (void)out_size; (void)ws_size;

    const dim3 blk(256);
    const dim3 gemmGrid(NN / 32, CC / 128);

    // degrees
    k_zero<<<NN / 256, blk, 0, stream>>>(deg, NN);
    k_degree<<<EE / 256, blk, 0, stream>>>(dstI, deg);
    k_rdeg<<<NN / 256, blk, 0, stream>>>(deg, rdeg);

    // ---- layer 0 (K = DIN = 128) ----
    k_zero<<<(int)((long)NN * DIN / 256), blk, 0, stream>>>(big0, (long)NN * DIN);
    k_scatter<<<(int)((long)EE * (DIN / 4) / 256), blk, 0, stream>>>(x, srcI, dstI, big0, DIN);
    k_gemm<<<gemmGrid, blk, 0, stream>>>(big0, l0_ll_w, rdeg, x, l0_lr_w, l0_ll_b,
                                         nullptr, big1, NN, CC, DIN);
    k_rownorm<<<NN / 8, blk, 0, stream>>>(big1);
    k_gemm<<<gemmGrid, blk, 0, stream>>>(x, l0_res_w, nullptr, nullptr, nullptr, nullptr,
                                         nullptr, big3, NN, CC, DIN);
    k_gmean<<<BB, blk, 0, stream>>>(big1, gmean);
    k_gvar<<<BB, blk, 0, stream>>>(big1, gmean, gn_a, gvar);
    k_gnapply<<<(int)(BIG / 256), blk, 0, stream>>>(big1, gmean, gvar, gn_w, gn_b, gn_a,
                                                    big3, big2);

    // ---- layers 1..3 (K = C = 256) ----
    for (int i = 0; i < 3; ++i) {
        const float* wl = lin_l_w + (size_t)i * CC * CC;
        const float* wr = lin_r_w + (size_t)i * CC * CC;
        const float* bl = lin_l_b + (size_t)i * CC;
        const float* gw = gn_w + (size_t)(i + 1) * CC;
        const float* gb = gn_b + (size_t)(i + 1) * CC;
        const float* ga = gn_a + (size_t)(i + 1) * CC;
        k_zero<<<(int)(BIG / 256), blk, 0, stream>>>(big0, (long)BIG);
        k_scatter<<<(int)((long)EE * (CC / 4) / 256), blk, 0, stream>>>(big2, srcI, dstI, big0, CC);
        k_gemm<<<gemmGrid, blk, 0, stream>>>(big0, wl, rdeg, big2, wr, bl, nullptr,
                                             big1, NN, CC, CC);
        k_rownorm<<<NN / 8, blk, 0, stream>>>(big1);
        k_gmean<<<BB, blk, 0, stream>>>(big1, gmean);
        k_gvar<<<BB, blk, 0, stream>>>(big1, gmean, ga, gvar);
        k_gnapply<<<(int)(BIG / 256), blk, 0, stream>>>(big1, gmean, gvar, gw, gb, ga,
                                                        big2, big2);
    }

    // ---- attention: q/k/v via sliced attn_in_w ----
    k_gemm<<<gemmGrid, blk, 0, stream>>>(big2, ain_w, nullptr, nullptr, nullptr,
                                         ain_b, nullptr, big0, NN, CC, CC);                 // q
    k_gemm<<<gemmGrid, blk, 0, stream>>>(big2, ain_w + (size_t)CC * CC, nullptr, nullptr,
                                         nullptr, ain_b + CC, nullptr, big3, NN, CC, CC);   // k
    k_gemm<<<gemmGrid, blk, 0, stream>>>(big2, ain_w + (size_t)2 * CC * CC, nullptr, nullptr,
                                         nullptr, ain_b + 2 * CC, nullptr, big4, NN, CC, CC); // v
    k_attn<<<BB * HH * (NP / 128), blk, 0, stream>>>(big0, big3, big4, big1);
    // out-proj + residual (x2 = heads @ Wo^T + bo + x1) -> big0
    k_gemm<<<gemmGrid, blk, 0, stream>>>(big1, aout_w, nullptr, nullptr, nullptr,
                                         aout_b, big2, big0, NN, CC, CC);

    // ---- graph LayerNorm + gelu ----
    k_lnstats<<<BB, blk, 0, stream>>>(big0, lnmu, lnvar);
    k_lnapply<<<(int)(BIG / 256), blk, 0, stream>>>(big0, lnmu, lnvar, ln_w, ln_b, big1);

    // ---- pooling + final projection ----
    k_pool<<<BB, blk, 0, stream>>>(big1, pooled);
    k_gemm<<<dim3(BB / 32, CC / 128), blk, 0, stream>>>(pooled, pool_w, nullptr, nullptr,
                                                        nullptr, pool_b, nullptr, out,
                                                        BB, CC, 768);
}
